// mamba_not_h3_59158879535674
// MI455X (gfx1250) — compile-verified, matrix operand order checked by reading, not run on hardware
//
#include <hip/hip_runtime.h>
#include <math.h>

// ---------------------------------------------------------------- types ----
typedef __attribute__((ext_vector_type(16))) _Float16 v16h;
typedef __attribute__((ext_vector_type(8)))  _Float16 half8;
typedef __attribute__((ext_vector_type(8)))  float    v8f;

#define BATCH 8
#define SEQ   512          // sequence length (= IN_CH of conv stem)
#define DI    1024         // d_inner
#define NST   16           // state size
#define RDT   32           // dt_rank

__device__ __forceinline__ float fsilu(float x) { return x / (1.f + __expf(-x)); }

// Load one 16-bit WMMA operand fragment (A row / B column) as two b128 loads.
// Element packing per CDNA5 ISA 7.12.2: elems 0..7 -> K = half*8 .. +7,
// elems 8..15 -> K = 16 + half*8 .. +7  (p already offset by half*8).
__device__ __forceinline__ v16h load_frag16(const _Float16* __restrict__ p) {
    half8 lo = *(const half8*)(p);
    half8 hi = *(const half8*)(p + 16);
    return __builtin_shufflevector(lo, hi, 0,1,2,3,4,5,6,7,8,9,10,11,12,13,14,15);
}

// ------------------------------------------------ generic WMMA GEMM -------
// D(M,N) = A(M,K) f16 * B(N,K)^T f16, batched via strides (elements).
// Grid: (N/64, M/64, batch). Block: 128 threads = 4 waves; wave -> 16x64 strip.
// Software-pipelined: next K-slice fragments load while current slice's
// 4 WMMAs issue. Epilogue fused via MODE:
//   0: f32 store
//   1: f16 store
//   2: conv-stem: +e0[m]; BN(e1..e4)[m]; SiLU; 2x + PE(m,n); f16 store
//   3: softplus(x + e0[n]); f32 store
template<int MODE>
__global__ __launch_bounds__(128) void wmma_gemm(
    const _Float16* __restrict__ A, const _Float16* __restrict__ B,
    void* __restrict__ Dp, int K, int lda, int ldb, int ldd,
    long long sA, long long sB, long long sD,
    const float* __restrict__ e0, const float* __restrict__ e1,
    const float* __restrict__ e2, const float* __restrict__ e3,
    const float* __restrict__ e4)
{
    const int lane = threadIdx.x & 31;
    const int wave = threadIdx.x >> 5;
    A += (long long)blockIdx.z * sA;
    B += (long long)blockIdx.z * sB;

    const int m0 = blockIdx.y * 64 + wave * 16;
    const int n0 = blockIdx.x * 64;
    const int r  = lane & 15;        // row of A / col of B held by this lane
    const int hf = lane >> 4;        // lane half selects K sub-block

    const _Float16* ap = A + (size_t)(m0 + r) * lda + (hf << 3);
    const _Float16* bp = B + (size_t)(n0 + r) * ldb + (hf << 3);
    const size_t bstep = (size_t)16 * ldb;

    v8f acc[4] = {v8f{}, v8f{}, v8f{}, v8f{}};

    // prologue: slice 0 fragments
    v16h a  = load_frag16(ap);
    v16h b0 = load_frag16(bp);
    v16h b1 = load_frag16(bp + bstep);
    v16h b2 = load_frag16(bp + 2 * bstep);
    v16h b3 = load_frag16(bp + 3 * bstep);

    for (int k0 = 32; k0 < K; k0 += 32) {
        __builtin_prefetch(ap + k0 + 32, 0, 3);       // global_prefetch_b8
        __builtin_prefetch(bp + k0 + 32, 0, 3);
        v16h an = load_frag16(ap + k0);               // next slice -> fresh regs
        v16h c0 = load_frag16(bp + k0);
        v16h c1 = load_frag16(bp + k0 + bstep);
        v16h c2 = load_frag16(bp + k0 + 2 * bstep);
        v16h c3 = load_frag16(bp + k0 + 3 * bstep);
        acc[0] = __builtin_amdgcn_wmma_f32_16x16x32_f16(false, a, false, b0, (short)0, acc[0], false, false);
        acc[1] = __builtin_amdgcn_wmma_f32_16x16x32_f16(false, a, false, b1, (short)0, acc[1], false, false);
        acc[2] = __builtin_amdgcn_wmma_f32_16x16x32_f16(false, a, false, b2, (short)0, acc[2], false, false);
        acc[3] = __builtin_amdgcn_wmma_f32_16x16x32_f16(false, a, false, b3, (short)0, acc[3], false, false);
        a = an; b0 = c0; b1 = c1; b2 = c2; b3 = c3;
    }
    acc[0] = __builtin_amdgcn_wmma_f32_16x16x32_f16(false, a, false, b0, (short)0, acc[0], false, false);
    acc[1] = __builtin_amdgcn_wmma_f32_16x16x32_f16(false, a, false, b1, (short)0, acc[1], false, false);
    acc[2] = __builtin_amdgcn_wmma_f32_16x16x32_f16(false, a, false, b2, (short)0, acc[2], false, false);
    acc[3] = __builtin_amdgcn_wmma_f32_16x16x32_f16(false, a, false, b3, (short)0, acc[3], false, false);

    // C/D layout: VGPR i holds M = m0 + hf*8 + i, N = n0 + j*16 + (lane&15)
    const int gm0 = m0 + hf * 8;
    #pragma unroll
    for (int i = 0; i < 8; ++i) {
        const int gm = gm0 + i;
        float rowadd = 0.f, rowmul = 1.f;
        if (MODE == 2) {
            // fold bias + BN into one mul/add pair for this output row
            float scale = e1[gm] * rsqrtf(e4[gm] + 1e-5f);
            rowmul = scale;
            rowadd = (e0[gm] - e3[gm]) * scale + e2[gm];
        }
        #pragma unroll
        for (int j = 0; j < 4; ++j) {
            const int gn = n0 + 16 * j + r;
            float v = acc[j][i];
            size_t idx = (size_t)gm * ldd + gn;
            if (MODE == 0) {
                ((float*)Dp + (long long)blockIdx.z * sD)[idx] = v;
            } else if (MODE == 1) {
                ((_Float16*)Dp + (long long)blockIdx.z * sD)[idx] = (_Float16)v;
            } else if (MODE == 2) {
                v = v * rowmul + rowadd;                 // +pw_b then BN(eval)
                v = 2.f * fsilu(v);
                float ang = (float)gm * __expf((float)(gn & ~1) * (-9.210340371976184f / 512.f));
                v += (gn & 1) ? __cosf(ang) : __sinf(ang);
                ((_Float16*)Dp + (long long)blockIdx.z * sD)[idx] = (_Float16)v;
            } else {                                     // MODE 3: softplus
                float x = v + e0[gn];
                ((float*)Dp + (long long)blockIdx.z * sD)[idx] =
                    (x > 20.f) ? x : log1pf(__expf(x));
            }
        }
    }
}

// ------------------------------------------------ elementwise kernels -----
__global__ void cvt_f32_f16(const float* __restrict__ s, _Float16* __restrict__ d, int n) {
    int i = blockIdx.x * blockDim.x + threadIdx.x;
    if (i < n) d[i] = (_Float16)s[i];
}

// depthwise k=2 s=2; writes transposed (b, l, c) so pointwise GEMM sees (N,K)
__global__ void dw_stem(const float* __restrict__ x, const float* __restrict__ dww,
                        const float* __restrict__ dwb, _Float16* __restrict__ tT) {
    int i = blockIdx.x * blockDim.x + threadIdx.x;   // ((b*SEQ + l)*512 + c)
    int c = i & 511, l = (i >> 9) & 511, b = i >> 18;
    const float* xp = x + ((size_t)b * 512 + c) * 1024 + 2 * l;
    tT[i] = (_Float16)(xp[0] * dww[2 * c] + xp[1] * dww[2 * c + 1] + dwb[c]);
}

// causal depthwise conv k=4 over seq + SiLU; f32 copy for scan, f16 for x_proj
__global__ void causal_conv(const _Float16* __restrict__ xz16, const float* __restrict__ w,
                            const float* __restrict__ bias, float* __restrict__ xcf,
                            _Float16* __restrict__ xch) {
    int i = blockIdx.x * blockDim.x + threadIdx.x;   // ((b*SEQ + l)*DI + d)
    int d = i & 1023, l = (i >> 10) & 511, b = i >> 19;
    const _Float16* base = xz16 + (size_t)b * SEQ * 2048 + d;   // x_in half of xz
    float acc = bias[d];
    #pragma unroll
    for (int k = 0; k < 4; ++k) {
        int ls = l - 3 + k;
        if (ls >= 0) acc += (float)base[(size_t)ls * 2048] * w[d * 4 + k];
    }
    float v = fsilu(acc);
    xcf[i] = v;
    xch[i] = (_Float16)v;
}

__global__ void take_dt(const float* __restrict__ xdbl, _Float16* __restrict__ dt16) {
    int i = blockIdx.x * blockDim.x + threadIdx.x;   // ((b*SEQ + l)*32 + r)
    int r = i & 31, bl = i >> 5;
    dt16[i] = (_Float16)xdbl[(size_t)bl * 64 + r];
}

// selective scan: thread per (b,d); 16 states in registers; fused epilogue
__global__ __launch_bounds__(256) void sel_scan(
    const float* __restrict__ dt, const float* __restrict__ xdbl,
    const float* __restrict__ xc, const _Float16* __restrict__ xz16,
    const float* __restrict__ A_log, const float* __restrict__ Dvec,
    _Float16* __restrict__ yact) {
    int tid = blockIdx.x * blockDim.x + threadIdx.x;   // b*DI + d
    int d = tid & (DI - 1), b = tid >> 10;
    float a[NST], h[NST];
    #pragma unroll
    for (int n = 0; n < NST; ++n) { a[n] = -__expf(A_log[d * NST + n]); h[n] = 0.f; }
    float Dd = Dvec[d];
    for (int l = 0; l < SEQ; ++l) {
        size_t bl = (size_t)b * SEQ + l;
        float dtv = dt[bl * DI + d];
        float xcv = xc[bl * DI + d];
        const float* xd = xdbl + bl * 64;
        float y = 0.f;
        #pragma unroll
        for (int n = 0; n < NST; ++n) {
            float dA = __expf(dtv * a[n]);
            h[n] = dA * h[n] + (dtv * xd[32 + n]) * xcv;
            y += h[n] * xd[48 + n];
        }
        y += xcv * Dd;
        float z = (float)xz16[bl * 2048 + 1024 + d];
        y *= fsilu(z);
        yact[bl * DI + d] = (_Float16)y;
    }
}

// ---------------------------------------------------------------- launch --
extern "C" void kernel_launch(void* const* d_in, const int* in_sizes, int n_in,
                              void* d_out, int out_size, void* d_ws, size_t ws_size,
                              hipStream_t stream) {
    const float* x     = (const float*)d_in[0];
    const float* dw_w  = (const float*)d_in[1];
    const float* dw_b  = (const float*)d_in[2];
    const float* pw_w  = (const float*)d_in[3];
    const float* pw_b  = (const float*)d_in[4];
    const float* bn_g  = (const float*)d_in[5];
    const float* bn_b  = (const float*)d_in[6];
    const float* bn_m  = (const float*)d_in[7];
    const float* bn_v  = (const float*)d_in[8];
    const float* ipw   = (const float*)d_in[9];
    const float* c1w   = (const float*)d_in[10];
    const float* c1b   = (const float*)d_in[11];
    const float* xpw   = (const float*)d_in[12];
    const float* dtw   = (const float*)d_in[13];
    const float* dtb   = (const float*)d_in[14];
    const float* Alog  = (const float*)d_in[15];
    const float* Dv    = (const float*)d_in[16];
    const float* opw   = (const float*)d_in[17];
    float* out = (float*)d_out;

    // -------- workspace carving (256B aligned) --------
    char* ws = (char*)d_ws;
    size_t off = 0;
    auto carve = [&](size_t bytes) { char* p = ws + off; off = (off + bytes + 255) & ~(size_t)255; return p; };
    _Float16* tT16   = (_Float16*)carve((size_t)BATCH * SEQ * 512 * 2);   // dw out, (b,l,c)
    _Float16* hseq16 = (_Float16*)carve((size_t)BATCH * 512 * 512 * 2);   // stem out (fused epi)
    _Float16* xz16   = (_Float16*)carve((size_t)BATCH * SEQ * 2048 * 2);  // in_proj out, f16
    float*    xcf    = (float*)   carve((size_t)BATCH * SEQ * DI * 4);
    _Float16* xch    = (_Float16*)carve((size_t)BATCH * SEQ * DI * 2);
    float*    xdbl   = (float*)   carve((size_t)BATCH * SEQ * 64 * 4);
    _Float16* dt16   = (_Float16*)carve((size_t)BATCH * SEQ * RDT * 2);
    float*    dtf    = (float*)   carve((size_t)BATCH * SEQ * DI * 4);    // softplus(dt) (fused epi)
    _Float16* yact16 = (_Float16*)carve((size_t)BATCH * SEQ * DI * 2);
    _Float16* pw16   = (_Float16*)carve((size_t)512 * 512 * 2);
    _Float16* ip16   = (_Float16*)carve((size_t)2048 * 512 * 2);
    _Float16* xp16   = (_Float16*)carve((size_t)64 * 1024 * 2);
    _Float16* dtp16  = (_Float16*)carve((size_t)1024 * 32 * 2);
    _Float16* op16   = (_Float16*)carve((size_t)512 * 1024 * 2);
    (void)ws_size; (void)n_in; (void)in_sizes; (void)out_size;

    const int T = 256;
    auto blocks = [&](int n) { return (n + T - 1) / T; };

    // weight conversions f32 -> f16
    cvt_f32_f16<<<blocks(512*512),  T, 0, stream>>>(pw_w, pw16, 512*512);
    cvt_f32_f16<<<blocks(2048*512), T, 0, stream>>>(ipw,  ip16, 2048*512);
    cvt_f32_f16<<<blocks(64*1024),  T, 0, stream>>>(xpw,  xp16, 64*1024);
    cvt_f32_f16<<<blocks(1024*32),  T, 0, stream>>>(dtw,  dtp16, 1024*32);
    cvt_f32_f16<<<blocks(512*1024), T, 0, stream>>>(opw,  op16, 512*1024);

    // stem: depthwise k=2 s=2 (transposed write)
    dw_stem<<<blocks(BATCH*SEQ*512), T, 0, stream>>>(x, dw_w, dw_b, tT16);

    // pointwise 512x512x512 GEMM per batch; fused +pw_b, BN, SiLU, 2x+PE -> f16
    wmma_gemm<2><<<dim3(8, 8, BATCH), 128, 0, stream>>>(
        pw16, tT16, hseq16, 512, 512, 512, 512,
        0, (long long)512*512, (long long)512*512,
        pw_b, bn_g, bn_b, bn_m, bn_v);

    // in_proj: (512 x 2048 x 512) -> f16 directly (halves xz traffic)
    wmma_gemm<1><<<dim3(32, 8, BATCH), 128, 0, stream>>>(
        hseq16, ip16, xz16, 512, 512, 512, 2048,
        (long long)512*512, 0, (long long)512*2048,
        nullptr, nullptr, nullptr, nullptr, nullptr);

    causal_conv<<<blocks(BATCH*SEQ*DI), T, 0, stream>>>(xz16, c1w, c1b, xcf, xch);

    // x_proj: (512 x 64 x 1024) -> f32
    wmma_gemm<0><<<dim3(1, 8, BATCH), 128, 0, stream>>>(
        xch, xp16, xdbl, 1024, 1024, 1024, 64,
        (long long)512*DI, 0, (long long)512*64,
        nullptr, nullptr, nullptr, nullptr, nullptr);

    take_dt<<<blocks(BATCH*SEQ*RDT), T, 0, stream>>>(xdbl, dt16);

    // dt_proj: (512 x 1024 x 32); fused +dt_proj_b and softplus -> f32
    wmma_gemm<3><<<dim3(16, 8, BATCH), 128, 0, stream>>>(
        dt16, dtp16, dtf, 32, 32, 32, 1024,
        (long long)512*RDT, 0, (long long)512*DI,
        dtb, nullptr, nullptr, nullptr, nullptr);

    // selective scan (fused +xc*D and *silu(z))
    sel_scan<<<BATCH*DI/256, 256, 0, stream>>>(dtf, xdbl, xcf, xz16, Alog, Dv, yact16);

    // out_proj: (512 x 512 x 1024) -> d_out (f32)
    wmma_gemm<0><<<dim3(8, 8, BATCH), 128, 0, stream>>>(
        yact16, op16, out, 1024, 1024, 1024, 512,
        (long long)512*DI, 0, (long long)512*512,
        nullptr, nullptr, nullptr, nullptr, nullptr);
}
